// WinAtten_8280696947136
// MI455X (gfx1250) — matrix operand order repaired and checked by reading, not compiled, not run on hardware
//
#include <hip/hip_runtime.h>
#include <hip/hip_bf16.h>

// ---------------------------------------------------------------------------
// Windowed attention (denoiser) for MI455X / gfx1250, wave32 + WMMA f16.
//   B=2, H=W=512, CH=64, HEADS=4, HD=16, BLOCK=8 -> 8192 windows of 64 tokens.
// One workgroup = one 8x8 window; 128 threads = 4 wave32, one wave per head.
// All matmuls via v_wmma_f32_16x16x32_f16 (f32 accumulate), softmax in f32.
// Input staging uses GLOBAL_LOAD_ASYNC_TO_LDS (ASYNCcnt) when available.
// ---------------------------------------------------------------------------

typedef __attribute__((ext_vector_type(16))) _Float16 v16h;
typedef __attribute__((ext_vector_type(8)))  float    v8f;
typedef __attribute__((ext_vector_type(4)))  int      v4i;

#if defined(__HIP_DEVICE_COMPILE__) && \
    __has_builtin(__builtin_amdgcn_global_load_async_to_lds_b128) && \
    __has_builtin(__builtin_amdgcn_s_wait_asynccnt)
#define HAVE_ASYNC_LDS 1
typedef v4i __attribute__((address_space(1)))* g4i_t;   // global int4*
typedef v4i __attribute__((address_space(3)))* l4i_t;   // LDS int4*
#else
#define HAVE_ASYNC_LDS 0
#endif

#define N_HEADS   4
#define TOKENS    64
#define IMG       512
#define N_WIN     (2 * 64 * 64)

// LDS pitches (elements), padded to dodge 64-bank conflicts, 16B-aligned rows
#define AP 40     // aux / noisy: [64 px][32 in-ch]  f16
#define WP 40     // weights:     [64 out-ch][32 in-ch] f16
#define QP 136    // Q/K:         [64 px][4 heads * 32 d (top 16 zero)] f16
#define VP 72     // V:           [64 ch][64 px] f16
#define SP 68     // sim:         per head [64 row][64 col] f32
#define OPITCH 68 // Obuf:        [64 ch][64 px] f32 (aliases Q buffer)

// byte offsets inside the static LDS block
#define OFF_AUXC 0
#define OFF_AUXP 5120
#define OFF_NOI  10240
#define OFF_WQ   15360
#define OFF_WK   20480
#define OFF_WV   25600
#define OFF_REL  30720        // relH[64] f32, relW[64] f32
#define OFF_Q    31232        // 17408 B  (later aliased by Obuf)
#define OFF_K    48640        // 17408 B
#define OFF_V    66048        // 9216 B
#define OFF_SIM  75264        // 4*64*68*4 = 69632 B
#define OFF_STG  144896       // raw f32 staging: [31 ch][64 px] = 7936 B
#define SMEM_BYTES 152832

union FragAB { v16h v; uint4 q[2]; };

// A-matrix fragment (16x32 f16): lane<16 holds row m0+lane, K chunks [0..7],[16..23];
// lane>=16 holds row m0+lane-16, K chunks [8..15],[24..31].  LDS is row-major,
// K contiguous -> two ds_load_b128 per fragment.
__device__ __forceinline__ v16h load_frag_A(const _Float16* base, int pitch,
                                            int m0, int k0, int hlf, int lr) {
  FragAB f;
  const char* row = (const char*)(base + (m0 + lr) * pitch + k0);
  f.q[0] = *(const uint4*)(row + hlf * 16);
  f.q[1] = *(const uint4*)(row + 32 + hlf * 16);
  return f.v;
}

// B-matrix fragment (32x16 f16): lane<16 = column n0+lane, K chunks [0..7],[16..23];
// lane>=16 = column n0+lane-16, K chunks [8..15],[24..31].  LDS rows indexed by N, K contiguous.
__device__ __forceinline__ v16h load_frag_B(const _Float16* base, int pitch,
                                            int n0, int k0, int hlf, int lr) {
  FragAB f;
  const char* row = (const char*)(base + (n0 + lr) * pitch + k0);
  f.q[0] = *(const uint4*)(row + hlf * 16);
  f.q[1] = *(const uint4*)(row + 32 + hlf * 16);
  return f.v;
}

__device__ __forceinline__ v8f wmma16(v16h a, v16h b, v8f c) {
  return __builtin_amdgcn_wmma_f32_16x16x32_f16(false, a, false, b,
                                                (short)0, c, false, false);
}

__global__ __launch_bounds__(128)
void winatten_kernel(const float* __restrict__ noisy,
                     const float* __restrict__ curr,
                     const float* __restrict__ prev,
                     const float* __restrict__ qw,
                     const float* __restrict__ kw,
                     const float* __restrict__ vw,
                     const float* __restrict__ relh,
                     const float* __restrict__ relw,
                     float* __restrict__ out) {
  __shared__ __align__(16) char smem[SMEM_BYTES];
  _Float16* auxC = (_Float16*)(smem + OFF_AUXC);
  _Float16* auxP = (_Float16*)(smem + OFF_AUXP);
  _Float16* noiL = (_Float16*)(smem + OFF_NOI);
  _Float16* Wq   = (_Float16*)(smem + OFF_WQ);
  _Float16* Wk   = (_Float16*)(smem + OFF_WK);
  _Float16* Wv   = (_Float16*)(smem + OFF_WV);
  float*    relH = (float*)(smem + OFF_REL);
  float*    relW = relH + 64;
  _Float16* Qlds = (_Float16*)(smem + OFF_Q);
  _Float16* Klds = (_Float16*)(smem + OFF_K);
  _Float16* Vlds = (_Float16*)(smem + OFF_V);
  float*    simf = (float*)(smem + OFF_SIM);
  float*    Obuf = (float*)(smem + OFF_Q);     // alias: Q dead after sim phase
  float*    stg  = (float*)(smem + OFF_STG);   // raw f32 window staging

  const int tid  = threadIdx.x;
  const int wave = tid >> 5;
  const int lane = tid & 31;
  const int hlf  = lane >> 4;   // lane-half selector for WMMA fragments
  const int lr   = lane & 15;

  const int win = blockIdx.x;
  const int bb  = win >> 12;            // batch
  const int wy  = (win >> 6) & 63;      // window row
  const int wx  = win & 63;             // window col
  const int py0 = wy * 8, px0 = wx * 8;

  // ---- Phase 0a: kick off async staging of window inputs into LDS --------
  // stg rows: c=0..13 curr_aux, 14..27 prev_aux, 28..30 noisy; [c][64 px] f32.
  // 31 ch x 8 rows x 2 x 16B lane-tasks = 496.
  for (int t = tid; t < 496; t += 128) {
    const int seg = t & 1, rr = (t >> 1) & 7, c = t >> 4;
    const float* srcb;
    int ch, nch;
    if (c < 14)      { srcb = curr;  ch = c;      nch = 14; }
    else if (c < 28) { srcb = prev;  ch = c - 14; nch = 14; }
    else             { srcb = noisy; ch = c - 28; nch = 3;  }
    const size_t s = (((size_t)(bb * nch + ch) * IMG + py0 + rr) * IMG + px0)
                     + (size_t)(seg * 4);
    float* dst = stg + (c * 64 + rr * 8 + seg * 4);
#if HAVE_ASYNC_LDS
    __builtin_amdgcn_global_load_async_to_lds_b128((g4i_t)(srcb + s),
                                                   (l4i_t)dst, 0, 0);
#else
    *(uint4*)dst = *(const uint4*)(srcb + s);
#endif
  }

  // ---- Phase 0b: zero K-padding regions (aux/weights + whole Q/K) --------
  {
    uint4 z = {0u, 0u, 0u, 0u};
    uint4* z0 = (uint4*)smem;                 // aux + weights + rel
    for (int i = tid; i < OFF_Q / 16; i += 128) z0[i] = z;
    uint4* z1 = (uint4*)(smem + OFF_Q);       // Q + K (34816 B)
    for (int i = tid; i < (OFF_V - OFF_Q) / 16; i += 128) z1[i] = z;
  }

  // ---- Phase 1a: stage weights and rel bias (tiny, direct) ----------------
  __syncthreads();
  for (int t = tid; t < 896; t += 128) {       // qw,kw: 64x14
    int o = t / 14, i = t % 14;
    Wq[o * WP + i] = (_Float16)qw[t];
    Wk[o * WP + i] = (_Float16)kw[t];
  }
  for (int t = tid; t < 192; t += 128) {       // vw: 64x3
    int o = t / 3, i = t % 3;
    Wv[o * WP + i] = (_Float16)vw[t];
  }
  if (tid < 64)        relH[tid]      = relh[tid];
  else if (tid < 128)  relW[tid - 64] = relw[tid - 64];

  // ---- Phase 1b: wait for async staging, then transpose/convert to f16 ---
#if HAVE_ASYNC_LDS
  __builtin_amdgcn_s_wait_asynccnt(0);
#endif
  __syncthreads();
  for (int t = tid; t < 31 * 8; t += 128) {    // (c, row) tasks
    const int rr = t & 7, c = t >> 3;
    const float* srow = stg + c * 64 + rr * 8;
    float4 a0 = *(const float4*)srow;
    float4 a1 = *(const float4*)(srow + 4);
    _Float16* dstb;
    int ch;
    if (c < 14)      { dstb = auxC; ch = c;      }
    else if (c < 28) { dstb = auxP; ch = c - 14; }
    else             { dstb = noiL; ch = c - 28; }
    _Float16* d = dstb + (rr * 8) * AP + ch;
    d[0*AP]=(_Float16)a0.x; d[1*AP]=(_Float16)a0.y; d[2*AP]=(_Float16)a0.z; d[3*AP]=(_Float16)a0.w;
    d[4*AP]=(_Float16)a1.x; d[5*AP]=(_Float16)a1.y; d[6*AP]=(_Float16)a1.z; d[7*AP]=(_Float16)a1.w;
  }
  __syncthreads();

  // ---- Phase 2: Q/K/V projections via WMMA (M=out-ch, N=px, K=in-ch pad32)
  // Straight-line, wave-striped: no divergent stores.
#pragma unroll
  for (int tt = 0; tt < 4; ++tt) {             // Q: tiles wave, wave+4, ...
    const int tile = wave + tt * 4;
    const int mi = tile >> 2, nj = tile & 3;
    v16h a = load_frag_A(Wq, WP, mi * 16, 0, hlf, lr);
    v16h b = load_frag_B(auxC, AP, nj * 16, 0, hlf, lr);
    v8f acc = {};
    acc = wmma16(a, b, acc);
    const int px = nj * 16 + lr;
    union { _Float16 h[8]; uint4 q; } pk;
#pragma unroll
    for (int r = 0; r < 8; ++r) pk.h[r] = (_Float16)(acc[r] * 0.25f);
    *(uint4*)(Qlds + px * QP + mi * 32 + hlf * 8) = pk.q;     // mi == head
  }
#pragma unroll
  for (int tt = 0; tt < 4; ++tt) {             // K: + rel-position bias
    const int tile = wave + tt * 4;
    const int mi = tile >> 2, nj = tile & 3;
    v16h a = load_frag_A(Wk, WP, mi * 16, 0, hlf, lr);
    v16h b = load_frag_B(auxP, AP, nj * 16, 0, hlf, lr);
    v8f acc = {};
    acc = wmma16(a, b, acc);
    const int px = nj * 16 + lr;
    const int bi = px >> 3, bj = px & 7;
    const float* bias = hlf ? (relW + bj * 8) : (relH + bi * 8);
    union { _Float16 h[8]; uint4 q; } pk;
#pragma unroll
    for (int r = 0; r < 8; ++r) pk.h[r] = (_Float16)(acc[r] + bias[r]);
    *(uint4*)(Klds + px * QP + mi * 32 + hlf * 8) = pk.q;
  }
#pragma unroll
  for (int tt = 0; tt < 4; ++tt) {             // V: [ch][px] layout
    const int tile = wave + tt * 4;
    const int mi = tile >> 2, nj = tile & 3;
    v16h a = load_frag_A(Wv, WP, mi * 16, 0, hlf, lr);
    v16h b = load_frag_B(noiL, AP, nj * 16, 0, hlf, lr);
    v8f acc = {};
    acc = wmma16(a, b, acc);
    const int px = nj * 16 + lr;
#pragma unroll
    for (int r = 0; r < 8; ++r)
      Vlds[(mi * 16 + hlf * 8 + r) * VP + px] = (_Float16)acc[r];
  }
  __syncthreads();

  // ---- Phase 3: sim = Q * K^T per head (M=N=64 px, K=16 pad32) ------------
  {
    const int h = wave;
    float* sim = simf + h * TOKENS * SP;
    const _Float16* Qh = Qlds + h * 32;
    const _Float16* Kh = Klds + h * 32;
#pragma unroll
    for (int mi = 0; mi < 4; ++mi) {
      v16h a = load_frag_A(Qh, QP, mi * 16, 0, hlf, lr);
#pragma unroll
      for (int nj = 0; nj < 4; ++nj) {
        v16h b = load_frag_B(Kh, QP, nj * 16, 0, hlf, lr);
        v8f acc = {};
        acc = wmma16(a, b, acc);
#pragma unroll
        for (int r = 0; r < 8; ++r)
          sim[(mi * 16 + hlf * 8 + r) * SP + nj * 16 + lr] = acc[r];
      }
    }
  }

  // ---- Phase 4: row softmax (f32), probs written f16 in-place -------------
  {
    const int h = wave;
    float* sim = simf + h * TOKENS * SP;
#pragma unroll
    for (int rr = 0; rr < 2; ++rr) {
      const int i = lane + rr * 32;          // each lane owns 2 rows of head h
      float* row = sim + i * SP;
      float4 v[16];
#pragma unroll
      for (int j = 0; j < 16; ++j) v[j] = *(const float4*)(row + j * 4);
      float m = v[0].x;
#pragma unroll
      for (int j = 0; j < 16; ++j)
        m = fmaxf(m, fmaxf(fmaxf(v[j].x, v[j].y), fmaxf(v[j].z, v[j].w)));
      float s = 0.f;
#pragma unroll
      for (int j = 0; j < 16; ++j) {
        v[j].x = __expf(v[j].x - m); v[j].y = __expf(v[j].y - m);
        v[j].z = __expf(v[j].z - m); v[j].w = __expf(v[j].w - m);
        s += v[j].x + v[j].y + v[j].z + v[j].w;
      }
      const float inv = 1.f / s;
      __asm__ __volatile__("" ::: "memory");  // order reads before alias write
      _Float16* prow = (_Float16*)row;        // P row aliases sim row start
#pragma unroll
      for (int j = 0; j < 16; ++j) {
        union { _Float16 h4[4]; uint2 u; } pk;
        pk.h4[0] = (_Float16)(v[j].x * inv); pk.h4[1] = (_Float16)(v[j].y * inv);
        pk.h4[2] = (_Float16)(v[j].z * inv); pk.h4[3] = (_Float16)(v[j].w * inv);
        *(uint2*)(prow + j * 4) = pk.u;
      }
    }
  }
  __syncthreads();   // also fences Q reads (sim) before Obuf alias writes

  // ---- Phase 5: out = P * V per head (M=64 px, N=16 d, K=64 in 2 steps) ---
  {
    const int h = wave;
    const _Float16* P = (const _Float16*)(simf + h * TOKENS * SP);
    const int PPITCH = SP * 2;               // 136 f16 elements per row
    const _Float16* Vh = Vlds + (h * 16) * VP;
    float* Ob = Obuf + (h * 16) * OPITCH;
#pragma unroll
    for (int mi = 0; mi < 4; ++mi) {
      v8f acc = {};
#pragma unroll
      for (int ks = 0; ks < 2; ++ks) {
        v16h a = load_frag_A(P, PPITCH, mi * 16, ks * 32, hlf, lr);
        v16h b = load_frag_B(Vh, VP, 0, ks * 32, hlf, lr);
        acc = wmma16(a, b, acc);
      }
      float* dst = Ob + lr * OPITCH + mi * 16 + hlf * 8;   // [ch][px], contig
#pragma unroll
      for (int r = 0; r < 8; ++r) dst[r] = acc[r];
    }
  }
  __syncthreads();

  // ---- Phase 6: coalesced global store (64 ch x 8 rows x 32B) -------------
  for (int t = tid; t < 512; t += 128) {
    const int ch = t >> 3, row = t & 7;
    const float* src = Obuf + ch * OPITCH + row * 8;
    float4 x0 = *(const float4*)(src);
    float4 x1 = *(const float4*)(src + 4);
    size_t d = ((size_t)(bb * 64 + ch) * IMG + (py0 + row)) * IMG + px0;
    *(float4*)(out + d)     = x0;
    *(float4*)(out + d + 4) = x1;
  }
}

extern "C" void kernel_launch(void* const* d_in, const int* in_sizes, int n_in,
                              void* d_out, int out_size, void* d_ws, size_t ws_size,
                              hipStream_t stream) {
  const float* noisy = (const float*)d_in[0];
  const float* curr  = (const float*)d_in[1];
  const float* prev  = (const float*)d_in[2];
  const float* qw    = (const float*)d_in[3];
  const float* kw    = (const float*)d_in[4];
  const float* vw    = (const float*)d_in[5];
  const float* relh  = (const float*)d_in[6];
  const float* relw  = (const float*)d_in[7];
  float* out = (float*)d_out;
  (void)in_sizes; (void)n_in; (void)out_size; (void)d_ws; (void)ws_size;
  winatten_kernel<<<N_WIN, 128, 0, stream>>>(noisy, curr, prev, qw, kw, vw,
                                             relh, relw, out);
}
